// Decoder_10325101379919
// MI455X (gfx1250) — matrix operand order repaired and checked by reading, not compiled, not run on hardware
//
#include <hip/hip_runtime.h>
#include <math.h>

// ---------------------------------------------------------------- types
typedef __attribute__((ext_vector_type(16))) _Float16 v16h;
typedef __attribute__((ext_vector_type(8)))  _Float16 v8h;
typedef __attribute__((ext_vector_type(8)))  float    v8f;

// ---------------------------------------------------------------- sizes
#define BB    32
#define TINN  512
#define EE    576
#define TOUTT 400
#define NMELL 80
#define PREE  256
#define ARNNN 1024
#define ADIMM 128
#define NFF   32
#define KSS   31

#define KATT  1856            // (PRE+E) + ARNN = 832 + 1024, 58 k-chunks of 32
#define KDEC  2624            // (ARNN+E) + DRNN = 1600 + 1024, 82 k-chunks
#define NG    4096            // 4*1024 gates
#define KT_ATT 58
#define KT_DEC 82
#define NWG   32
#define NTHR  256

// Gate-interleaved column permutation: packed col n -> original gate col.
// n = wg*128 + gate*32 + uu  <->  orig = gate*1024 + wg*32 + uu
__host__ __device__ inline int gate_perm(int n) {
  return ((n & 127) >> 5) * ARNNN + (n >> 7) * 32 + (n & 31);
}

// ---------------------------------------------------------------- WMMA helpers
__device__ inline v8f wmma_f16(v16h a, v16h b, v8f c) {
  return __builtin_amdgcn_wmma_f32_16x16x32_f16(false, a, false, b, (short)0, c,
                                                false, false);
}

// A fragment: 16x32 f16 tile from row-major [.,lda] activations.
__device__ inline v16h load_a_frag(const _Float16* A, int lda, int m0, int k0) {
  int lane = threadIdx.x & 31;
  const _Float16* p = A + (size_t)(m0 + (lane & 15)) * lda + k0 + ((lane >> 4) << 3);
  v8h lo = *(const v8h*)p;
  v8h hi = *(const v8h*)(p + 16);
  return __builtin_shufflevector(lo, hi, 0,1,2,3,4,5,6,7,8,9,10,11,12,13,14,15);
}

// B fragment: pre-swizzled so each lane reads 16 contiguous halves (32B).
__device__ inline v16h load_b_frag(const _Float16* Bsw, int ktiles, int kt, int ntile) {
  int lane = threadIdx.x & 31;
  const _Float16* p = Bsw + (((size_t)ntile * ktiles + kt) * 32 + lane) * 16;
  v8h lo = *(const v8h*)p;
  v8h hi = *(const v8h*)(p + 8);
  return __builtin_shufflevector(lo, hi, 0,1,2,3,4,5,6,7,8,9,10,11,12,13,14,15);
}

__device__ inline float sigf(float x) { return 1.f / (1.f + __expf(-x)); }

// ---------------------------------------------------------------- pack kernels
// Swizzle W[N][K-real] (row-major, out = x @ W.T) into B-fragment order for a
// (Ktot x N) f16 matrix; optional gate interleave; k >= K0+K1 -> zero pad.
__global__ void pack_bsw_kernel(_Float16* __restrict__ dst,
                                const float* __restrict__ W0, int K0,
                                const float* __restrict__ W1, int K1,
                                int Ktot, int N, int permute) {
  long total = (long)Ktot * N;
  int ktiles = Ktot >> 5;
  for (long idx = (long)blockIdx.x * blockDim.x + threadIdx.x; idx < total;
       idx += (long)gridDim.x * blockDim.x) {
    int j = (int)(idx & 15);
    long r = idx >> 4;
    int lane = (int)(r & 31); r >>= 5;
    int kt = (int)(r % ktiles);
    int nt = (int)(r / ktiles);
    int k = kt * 32 + ((lane >= 16) ? 8 : 0) + (j & 7) + ((j & 8) << 1);
    int n = nt * 16 + (lane & 15);
    int o = permute ? gate_perm(n) : n;
    float v = 0.f;
    if (k < K0)            v = W0[(long)o * K0 + k];
    else if (k < K0 + K1)  v = W1[(long)o * K1 + (k - K0)];
    dst[idx] = (_Float16)v;
  }
}

// di_f16[B][401][96]: go frame + teacher forcing, zero-padded 80->96
__global__ void pack_di_kernel(_Float16* __restrict__ dst,
                               const float* __restrict__ dec_in) {
  long total = (long)BB * (TOUTT + 1) * 96;
  for (long idx = (long)blockIdx.x * blockDim.x + threadIdx.x; idx < total;
       idx += (long)gridDim.x * blockDim.x) {
    int k = (int)(idx % 96);
    long r = idx / 96;
    int tt = (int)(r % (TOUTT + 1));
    int b  = (int)(r / (TOUTT + 1));
    float v = 0.f;
    if (k < NMELL && tt > 0)
      v = dec_in[((long)b * TOUTT + (tt - 1)) * NMELL + k];
    dst[idx] = (_Float16)v;
  }
}

__global__ void pack_f16_kernel(_Float16* __restrict__ dst,
                                const float* __restrict__ src, long n) {
  for (long i = (long)blockIdx.x * blockDim.x + threadIdx.x; i < n;
       i += (long)gridDim.x * blockDim.x)
    dst[i] = (_Float16)src[i];
}

__global__ void add_bias_kernel(float* __restrict__ dst,
                                const float* __restrict__ a,
                                const float* __restrict__ b, int n,
                                int permute) {
  int i = blockIdx.x * blockDim.x + threadIdx.x;
  if (i < n) {
    int o = permute ? gate_perm(i) : i;
    dst[i] = a[o] + b[o];
  }
}

// ---------------------------------------------------------------- generic WMMA GEMM
__global__ __launch_bounds__(32) void wmma_gemm_kernel(
    const _Float16* __restrict__ A, int lda,
    const _Float16* __restrict__ Bsw, int ktiles,
    float* __restrict__ Cf, _Float16* __restrict__ Ch, int ldc,
    int relu) {
  int ntile = blockIdx.x;
  int m0 = blockIdx.y * 16;
  v8f acc = {};
  for (int kt = 0; kt < ktiles; ++kt) {
    v16h a = load_a_frag(A, lda, m0, kt * 32);
    v16h b = load_b_frag(Bsw, ktiles, kt, ntile);
    acc = wmma_f16(a, b, acc);
  }
  int lane = threadIdx.x & 31;
  int n = ntile * 16 + (lane & 15);
  int mb = m0 + ((lane >> 4) << 3);
#pragma unroll
  for (int r = 0; r < 8; ++r) {
    float v = acc[r];
    if (relu) v = fmaxf(v, 0.f);
    size_t off = (size_t)(mb + r) * ldc + n;
    if (Cf) Cf[off] = v;
    if (Ch) Ch[off] = (_Float16)v;
  }
}

// ---------------------------------------------------------------- grid barrier
__device__ inline void gbar(unsigned* bar, unsigned* gen) {
  __syncthreads();
  if (threadIdx.x == 0) {
    unsigned g = *gen;
    unsigned arrived = __hip_atomic_fetch_add(&bar[0], 1u, __ATOMIC_ACQ_REL,
                                              __HIP_MEMORY_SCOPE_AGENT);
    if (arrived == gridDim.x - 1) {
      __hip_atomic_store(&bar[0], 0u, __ATOMIC_RELAXED, __HIP_MEMORY_SCOPE_AGENT);
      __hip_atomic_fetch_add(&bar[1], 1u, __ATOMIC_RELEASE, __HIP_MEMORY_SCOPE_AGENT);
    } else {
      while (__hip_atomic_load(&bar[1], __ATOMIC_ACQUIRE,
                               __HIP_MEMORY_SCOPE_AGENT) == g) {
        __builtin_amdgcn_s_sleep(1);
      }
    }
    *gen = g + 1;
  }
  __syncthreads();
}

// Gate GEMM into LDS: sG[32 batches][128 gate-interleaved cols] for this WG.
__device__ inline void lstm_gemm_lds(const _Float16* A, int lda, int ktiles,
                                     const _Float16* W, const float* bias,
                                     float* sG) {
  int wave = threadIdx.x >> 5;
  int ntile = blockIdx.x * 8 + wave;      // 0..255
  v8f acc0 = {};
  v8f acc1 = {};
  for (int kt = 0; kt < ktiles; ++kt) {
    v16h b  = load_b_frag(W, ktiles, kt, ntile);
    v16h a0 = load_a_frag(A, lda, 0,  kt * 32);
    v16h a1 = load_a_frag(A, lda, 16, kt * 32);
    acc0 = wmma_f16(a0, b, acc0);
    acc1 = wmma_f16(a1, b, acc1);
  }
  int lane = threadIdx.x & 31;
  int nl = wave * 16 + (lane & 15);                  // local col 0..127
  float bv = bias[blockIdx.x * 128 + nl];
  int mb = (lane >> 4) << 3;
#pragma unroll
  for (int r = 0; r < 8; ++r) {
    sG[(mb + r) * 128 + nl]      = acc0[r] + bv;
    sG[(mb + r + 16) * 128 + nl] = acc1[r] + bv;
  }
}

// ---------------------------------------------------------------- persistent scan
__global__ __launch_bounds__(NTHR, 1) void decoder_scan_kernel(
    const _Float16* __restrict__ xs,       // [B][401][256] prenet out (f16)
    const float* __restrict__ memory,      // [B][512][576]
    const float* __restrict__ pmem,        // [B][512][128]
    const _Float16* __restrict__ Watt, const float* __restrict__ batt,
    const _Float16* __restrict__ Wdec, const float* __restrict__ bdec,
    const float* __restrict__ qW,   const float* __restrict__ convW,
    const float* __restrict__ convb, const float* __restrict__ locW,
    const float* __restrict__ vW,
    const float* __restrict__ projW, const float* __restrict__ projb,
    const float* __restrict__ gateW, const float* __restrict__ gateb,
    const int* __restrict__ memlen,
    float* ah, float* ac, float* dh, float* dc,
    float* aw, float* awc, float* ctx,
    _Float16* attIn, _Float16* decIn,
    float* out_mel, float* out_gate, float* out_align,
    unsigned* bar) {
  __shared__ float sG[BB * 128];             // 16KB: gate slab (phases 1,3)
  __shared__ float s_ah[ARNNN];              // phase 2
  __shared__ float s_q[ADIMM];
  __shared__ float s_awh[TINN + KSS - 1];    // 542, halo of 15 each side
  __shared__ float s_awch[TINN + KSS - 1];
  __shared__ float s_en[TINN];
  __shared__ float s_red[NTHR];

  const int tid  = threadIdx.x;
  const int wg   = blockIdx.x;
  const int gtid = wg * NTHR + tid;
  const int gsz  = NWG * NTHR;
  unsigned gen = 0;

  // ---- init: attIn = [x_0 | 0 | 0], decIn = 0 -----------------------------
  for (int id = gtid; id < BB * KATT; id += gsz) {
    int b = id / KATT, k = id - b * KATT;
    attIn[id] = (k < PREE) ? xs[(size_t)b * (TOUTT + 1) * PREE + k]
                           : (_Float16)0.f;
  }
  for (int id = gtid; id < BB * KDEC; id += gsz) decIn[id] = (_Float16)0.f;
  gbar(bar, &gen);

  for (int t = 0; t < TOUTT; ++t) {
    // ==== Phase 1: attention LSTM (WMMA -> LDS -> fused pointwise) =========
    lstm_gemm_lds(attIn, KATT, KT_ATT, Watt, batt, sG);
    __syncthreads();
    for (int e = tid; e < BB * 32; e += NTHR) {
      int b = e >> 5, uu = e & 31;
      int u = (wg << 5) + uu;
      const float* g = sG + b * 128;
      float c = sigf(g[32 + uu]) * ac[b * ARNNN + u] +
                sigf(g[uu]) * tanhf(g[64 + uu]);
      ac[b * ARNNN + u] = c;
      ah[b * ARNNN + u] = c;            // source quirk: hidden := cell
      _Float16 ch = (_Float16)c;
      attIn[(size_t)b * KATT + PREE + EE + u] = ch;   // next step's ah input
      decIn[(size_t)b * KDEC + u] = ch;               // this step's dec input
    }
    gbar(bar, &gen);

    // ==== Phase 2: location-sensitive attention, one WG per batch row ======
    {
      const int b = wg;
      for (int u = tid; u < ARNNN; u += NTHR) s_ah[u] = ah[b * ARNNN + u];
      for (int p = tid; p < TINN + KSS - 1; p += NTHR) {
        int pp = p - (KSS / 2);
        bool ok = (pp >= 0 && pp < TINN);
        s_awh[p]  = ok ? aw[b * TINN + pp]  : 0.f;
        s_awch[p] = ok ? awc[b * TINN + pp] : 0.f;
      }
      __syncthreads();
      if (tid < ADIMM) {                       // q = ah @ qW.T
        float s = 0.f;
        const float* w = qW + (size_t)tid * ARNNN;
        for (int u = 0; u < ARNNN; ++u) s += s_ah[u] * w[u];
        s_q[tid] = s;
      }
      __syncthreads();
      const int mlen = memlen[b];
      for (int p = tid; p < TINN; p += NTHR) { // conv + energies
        float locf[NFF];
#pragma unroll 4
        for (int f = 0; f < NFF; ++f) {
          const float* w0 = convW + (size_t)f * 2 * KSS;
          float s = convb[f];
          for (int k = 0; k < KSS; ++k)
            s += w0[k] * s_awh[p + k] + w0[KSS + k] * s_awch[p + k];
          locf[f] = s;
        }
        float en = 0.f;
        const float* pm = pmem + ((size_t)b * TINN + p) * ADIMM;
        for (int a = 0; a < ADIMM; ++a) {
          const float* lw = locW + (size_t)a * NFF;
          float s = s_q[a] + pm[a];
          for (int f = 0; f < NFF; ++f) s += lw[f] * locf[f];
          en += tanhf(s) * vW[a];
        }
        if (p >= mlen) en = -INFINITY;
        s_en[p] = en;
      }
      __syncthreads();
      // softmax over 512
      s_red[tid] = fmaxf(s_en[tid], s_en[tid + NTHR]);
      __syncthreads();
      for (int s = NTHR / 2; s > 0; s >>= 1) {
        if (tid < s) s_red[tid] = fmaxf(s_red[tid], s_red[tid + s]);
        __syncthreads();
      }
      float mx = s_red[0];
      __syncthreads();
      float e0 = __expf(s_en[tid] - mx);
      float e1 = __expf(s_en[tid + NTHR] - mx);
      s_en[tid] = e0; s_en[tid + NTHR] = e1;
      s_red[tid] = e0 + e1;
      __syncthreads();
      for (int s = NTHR / 2; s > 0; s >>= 1) {
        if (tid < s) s_red[tid] += s_red[tid + s];
        __syncthreads();
      }
      float inv = 1.f / s_red[0];
      __syncthreads();
      for (int p = tid; p < TINN; p += NTHR) {
        float w = s_en[p] * inv;
        s_en[p] = w;
        aw[b * TINN + p]  = w;
        awc[b * TINN + p] = s_awch[KSS / 2 + p] + w;
        out_align[((size_t)b * TOUTT + t) * TINN + p] = w;
      }
      __syncthreads();
      for (int e = tid; e < EE; e += NTHR) {   // ctx = aw @ memory
        float s = 0.f;
        const float* mb = memory + (size_t)b * TINN * EE + e;
        for (int p = 0; p < TINN; ++p) s += s_en[p] * mb[(size_t)p * EE];
        ctx[b * EE + e] = s;
        _Float16 hh = (_Float16)s;
        attIn[(size_t)b * KATT + PREE + e] = hh;        // next step
        decIn[(size_t)b * KDEC + ARNNN + e] = hh;       // this step
      }
    }
    gbar(bar, &gen);

    // ==== Phase 3: decoder LSTM (WMMA -> LDS -> fused pointwise) ===========
    lstm_gemm_lds(decIn, KDEC, KT_DEC, Wdec, bdec, sG);
    __syncthreads();
    for (int e = tid; e < BB * 32; e += NTHR) {
      int b = e >> 5, uu = e & 31;
      int u = (wg << 5) + uu;
      const float* g = sG + b * 128;
      float c = sigf(g[32 + uu]) * dc[b * ARNNN + u] +
                sigf(g[uu]) * tanhf(g[64 + uu]);
      dc[b * ARNNN + u] = c;
      float h = sigf(g[96 + uu]) * tanhf(c);
      dh[b * ARNNN + u] = h;
      decIn[(size_t)b * KDEC + ARNNN + EE + u] = (_Float16)h;  // next step
    }
    if (t + 1 < TOUTT)                          // stage x_{t+1} (256 h / WG)
      attIn[(size_t)wg * KATT + tid] =
          xs[((size_t)wg * (TOUTT + 1) + t + 1) * PREE + tid];
    gbar(bar, &gen);

    // ==== Phase 4: projection + stop gate (no trailing barrier needed) =====
    for (int idx = gtid; idx < BB * (NMELL + 1); idx += gsz) {
      int b = idx / (NMELL + 1), m = idx - b * (NMELL + 1);
      const float* w = (m < NMELL) ? projW + (size_t)m * (ARNNN + EE) : gateW;
      float s = (m < NMELL) ? projb[m] : gateb[0];
      const float* hh = dh + (size_t)b * ARNNN;
      const float* cc = ctx + (size_t)b * EE;
      for (int u = 0; u < ARNNN; ++u) s += hh[u] * w[u];
      for (int e = 0; e < EE; ++e)    s += cc[e] * w[ARNNN + e];
      if (m < NMELL) out_mel[((size_t)b * NMELL + m) * TOUTT + t] = s;
      else           out_gate[(size_t)b * TOUTT + t] = s;
    }
    // Next phase 1 only writes ah/ac/attIn[ah]/decIn[ah]; phase 4 reads
    // dh/ctx -> no conflict, barrier elided.
  }
}

// ---------------------------------------------------------------- launch
extern "C" void kernel_launch(void* const* d_in, const int* in_sizes, int n_in,
                              void* d_out, int out_size, void* d_ws,
                              size_t ws_size, hipStream_t stream) {
  const float* memory  = (const float*)d_in[0];
  const float* dec_in  = (const float*)d_in[1];
  const float* W1      = (const float*)d_in[2];
  const float* W2      = (const float*)d_in[3];
  const float* attWih  = (const float*)d_in[4];
  const float* attWhh  = (const float*)d_in[5];
  const float* attbih  = (const float*)d_in[6];
  const float* attbhh  = (const float*)d_in[7];
  const float* qW      = (const float*)d_in[8];
  const float* memW    = (const float*)d_in[9];
  const float* convW   = (const float*)d_in[10];
  const float* convb   = (const float*)d_in[11];
  const float* locW    = (const float*)d_in[12];
  const float* vW      = (const float*)d_in[13];
  const float* decWih  = (const float*)d_in[14];
  const float* decWhh  = (const float*)d_in[15];
  const float* decbih  = (const float*)d_in[16];
  const float* decbhh  = (const float*)d_in[17];
  const float* projW   = (const float*)d_in[18];
  const float* projb   = (const float*)d_in[19];
  const float* gateW   = (const float*)d_in[20];
  const float* gateb   = (const float*)d_in[21];
  const int*   memlen  = (const int*)d_in[22];
  float* out = (float*)d_out;
  float* out_mel   = out;
  float* out_gate  = out + (size_t)BB * NMELL * TOUTT;
  float* out_align = out_gate + (size_t)BB * TOUTT;

  char* ws = (char*)d_ws;
  size_t off = 0;
  auto alloc = [&](size_t bytes) -> char* {
    char* p = ws + off;
    off = (off + bytes + 255) & ~(size_t)255;
    return p;
  };

  // state region (zero-initialized each call; barrier counters included)
  float* ah   = (float*)alloc(sizeof(float) * BB * ARNNN);
  float* ac   = (float*)alloc(sizeof(float) * BB * ARNNN);
  float* dh   = (float*)alloc(sizeof(float) * BB * ARNNN);
  float* dc   = (float*)alloc(sizeof(float) * BB * ARNNN);
  float* aw   = (float*)alloc(sizeof(float) * BB * TINN);
  float* awc  = (float*)alloc(sizeof(float) * BB * TINN);
  float* ctx  = (float*)alloc(sizeof(float) * BB * EE);
  unsigned* bar = (unsigned*)alloc(256);
  size_t state_bytes = off;

  _Float16*  attIn  = (_Float16*)alloc(sizeof(_Float16) * BB * KATT);
  _Float16*  decIn  = (_Float16*)alloc(sizeof(_Float16) * BB * KDEC);
  _Float16*  di16   = (_Float16*)alloc(sizeof(_Float16) * BB * (TOUTT + 1) * 96);
  _Float16*  pre1   = (_Float16*)alloc(sizeof(_Float16) * BB * (TOUTT + 1) * PREE);
  _Float16*  xs16   = (_Float16*)alloc(sizeof(_Float16) * BB * (TOUTT + 1) * PREE);
  _Float16*  mem16  = (_Float16*)alloc(sizeof(_Float16) * BB * TINN * EE);
  float*     pmem   = (float*)alloc(sizeof(float) * BB * TINN * ADIMM);
  _Float16*  Watt   = (_Float16*)alloc(sizeof(_Float16) * (size_t)KATT * NG);
  _Float16*  Wdec   = (_Float16*)alloc(sizeof(_Float16) * (size_t)KDEC * NG);
  _Float16*  W1sw   = (_Float16*)alloc(sizeof(_Float16) * 96 * PREE);
  _Float16*  W2sw   = (_Float16*)alloc(sizeof(_Float16) * PREE * PREE);
  _Float16*  memWsw = (_Float16*)alloc(sizeof(_Float16) * EE * ADIMM);
  float*     batt   = (float*)alloc(sizeof(float) * NG);
  float*     bdec   = (float*)alloc(sizeof(float) * NG);
  (void)ws_size; (void)in_sizes; (void)n_in; (void)out_size;

  hipMemsetAsync(ws, 0, state_bytes, stream);

  // ---- packing (one pass per call; deterministic) -------------------------
  pack_di_kernel<<<256, 256, 0, stream>>>(di16, dec_in);
  pack_f16_kernel<<<256, 256, 0, stream>>>(mem16, memory, (long)BB * TINN * EE);
  pack_bsw_kernel<<<64, 256, 0, stream>>>(W1sw, W1, NMELL, nullptr, 0, 96, PREE, 0);
  pack_bsw_kernel<<<64, 256, 0, stream>>>(W2sw, W2, PREE, nullptr, 0, PREE, PREE, 0);
  pack_bsw_kernel<<<64, 256, 0, stream>>>(memWsw, memW, EE, nullptr, 0, EE, ADIMM, 0);
  pack_bsw_kernel<<<1024, 256, 0, stream>>>(Watt, attWih, PREE + EE,
                                            attWhh, ARNNN, KATT, NG, 1);
  pack_bsw_kernel<<<1024, 256, 0, stream>>>(Wdec, decWih, ARNNN + EE,
                                            decWhh, ARNNN, KDEC, NG, 1);
  add_bias_kernel<<<(NG + 255) / 256, 256, 0, stream>>>(batt, attbih, attbhh, NG, 1);
  add_bias_kernel<<<(NG + 255) / 256, 256, 0, stream>>>(bdec, decbih, decbhh, NG, 1);

  // ---- prenet (two WMMA GEMMs with fused relu, f16 out) -------------------
  const int MROWS = BB * (TOUTT + 1);      // 12832, /16 = 802
  wmma_gemm_kernel<<<dim3(PREE / 16, MROWS / 16), 32, 0, stream>>>(
      di16, 96, W1sw, 3, nullptr, pre1, PREE, 1);
  wmma_gemm_kernel<<<dim3(PREE / 16, MROWS / 16), 32, 0, stream>>>(
      pre1, PREE, W2sw, PREE / 32, nullptr, xs16, PREE, 1);

  // ---- pmem = memory @ memW.T (WMMA, f32 out) -----------------------------
  wmma_gemm_kernel<<<dim3(ADIMM / 16, (BB * TINN) / 16), 32, 0, stream>>>(
      mem16, EE, memWsw, EE / 32, pmem, nullptr, ADIMM, 0);

  // ---- persistent sequential decoder scan (3 grid barriers / step) --------
  decoder_scan_kernel<<<NWG, NTHR, 0, stream>>>(
      xs16, memory, pmem, Watt, batt, Wdec, bdec, qW, convW, convb, locW, vW,
      projW, projb, gateW, gateb, memlen, ah, ac, dh, dc, aw, awc, ctx, attIn,
      decIn, out_mel, out_gate, out_align, bar);
}
